// CausalSelfAttention_29128468201633
// MI455X (gfx1250) — matrix operand order repaired and checked by reading, not compiled, not run on hardware
//
#include <hip/hip_runtime.h>
#include <hip/hip_bf16.h>

typedef _Float16 half_t;
typedef __attribute__((ext_vector_type(16))) _Float16 v16h;
typedef __attribute__((ext_vector_type(8)))  float    v8f;

#define Bv  4
#define Tv  2048
#define Cv  1024
#define NHv 16
#define HDv 64
#define Mtot (Bv*Tv)   // 8192

union AFrag {
    v16h h;
    unsigned int u[8];
    unsigned short s[16];
};

// ---------------------------------------------------------------------------
// f32 -> f16 conversion (4 elems/thread)
// ---------------------------------------------------------------------------
__global__ void cvt_f32_f16(const float* __restrict__ src, half_t* __restrict__ dst, int n)
{
    int i = (blockIdx.x * blockDim.x + threadIdx.x) * 4;
    if (i + 3 < n) {
        const float4 f = *reinterpret_cast<const float4*>(src + i);
        unsigned short u0 = __builtin_bit_cast(unsigned short, (half_t)f.x);
        unsigned short u1 = __builtin_bit_cast(unsigned short, (half_t)f.y);
        unsigned short u2 = __builtin_bit_cast(unsigned short, (half_t)f.z);
        unsigned short u3 = __builtin_bit_cast(unsigned short, (half_t)f.w);
        uint2 v;
        v.x = (unsigned)u0 | ((unsigned)u1 << 16);
        v.y = (unsigned)u2 | ((unsigned)u3 << 16);
        *reinterpret_cast<uint2*>(dst + i) = v;
    }
}

// ---------------------------------------------------------------------------
// WMMA GEMM:  out[m,n] = sum_k A[m,k] * W[n,k] + bias[n]
// A: [M,K] f16 row-major, W: [N,K] f16 row-major.
// mode 0: outF[m*N+n] (f32)
// mode 1: Q  -> outH[((b*NH+h)*T + t)*HD + d], scaled by 1/sqrt(HD)
// mode 2: K  -> outH[((b*NH+h)*T + t)*HD + d]
// mode 3: V  -> outH[((b*NH+h)*HD + d)*T + t]   (transposed for PV)
// block = 128 threads (4 waves, 2x2), block tile 128x128, wave tile 64x64.
// Double-buffered LDS: one barrier per k-step, global loads overlap WMMAs.
// ---------------------------------------------------------------------------
#define LDS_STRIDE 20   // uints per row (32 halves + pad), 80B: 16B aligned

__launch_bounds__(128, 1)
__global__ void gemm_wmma(const half_t* __restrict__ A,
                          const half_t* __restrict__ W,
                          const float*  __restrict__ bias,
                          half_t* __restrict__ outH,
                          float*  __restrict__ outF,
                          int M, int N, int K, int mode)
{
    __shared__ __attribute__((aligned(16))) unsigned int As[2][128 * LDS_STRIDE];
    __shared__ __attribute__((aligned(16))) unsigned int Bs[2][128 * LDS_STRIDE];

    const int tid  = threadIdx.x;
    const int wave = tid >> 5;
    const int lane = tid & 31;
    const int mloc = lane & 15;
    const int g    = lane >> 4;
    const int wm   = wave >> 1;
    const int wn   = wave & 1;
    const int m0   = blockIdx.y * 128;
    const int n0   = blockIdx.x * 128;

    const uint4* baseA = reinterpret_cast<const uint4*>(A + (size_t)(m0 + tid) * K);
    const uint4* baseB = reinterpret_cast<const uint4*>(W + (size_t)(n0 + tid) * K);

    v8f acc[4][4] = {};
    uint4 ra0, ra1, ra2, ra3, rb0, rb1, rb2, rb3;

    // ---- prologue: stage k-step 0 into buffer 0 ----
    ra0 = baseA[0]; ra1 = baseA[1]; ra2 = baseA[2]; ra3 = baseA[3];
    rb0 = baseB[0]; rb1 = baseB[1]; rb2 = baseB[2]; rb3 = baseB[3];
    {
        uint4* dA = reinterpret_cast<uint4*>(&As[0][tid * LDS_STRIDE]);
        dA[0] = ra0; dA[1] = ra1; dA[2] = ra2; dA[3] = ra3;
        uint4* dB = reinterpret_cast<uint4*>(&Bs[0][tid * LDS_STRIDE]);
        dB[0] = rb0; dB[1] = rb1; dB[2] = rb2; dB[3] = rb3;
    }
    __syncthreads();

    int buf = 0;
    for (int k0 = 0; k0 < K; k0 += 32) {
        const bool hasNext = (k0 + 32 < K);
        if (hasNext) {
            const uint4* sA = baseA + ((k0 + 32) >> 3);   // 8 halves per uint4
            const uint4* sB = baseB + ((k0 + 32) >> 3);
            ra0 = sA[0]; ra1 = sA[1]; ra2 = sA[2]; ra3 = sA[3];
            rb0 = sB[0]; rb1 = sB[1]; rb2 = sB[2]; rb3 = sB[3];
        }

        // ---- A fragments for this wave's 4 row tiles ----
        AFrag a[4];
        #pragma unroll
        for (int i = 0; i < 4; i++) {
            const unsigned int* p = &As[buf][(wm * 64 + i * 16 + mloc) * LDS_STRIDE];
            #pragma unroll
            for (int v = 0; v < 8; v++) {
                int kk = (g << 4) + (v << 1); // half index
                a[i].u[v] = p[kk >> 1];
            }
        }
        // ---- B fragments, one at a time, 16 WMMAs total ----
        #pragma unroll
        for (int j = 0; j < 4; j++) {
            AFrag bf;
            const unsigned int* p = &Bs[buf][(wn * 64 + j * 16 + mloc) * LDS_STRIDE + (g << 3)];
            #pragma unroll
            for (int v = 0; v < 8; v++) bf.u[v] = p[v];
            #pragma unroll
            for (int i = 0; i < 4; i++) {
                acc[i][j] = __builtin_amdgcn_wmma_f32_16x16x32_f16(
                    false, a[i].h, false, bf.h, (short)0, acc[i][j], false, false);
            }
        }

        if (hasNext) {
            uint4* dA = reinterpret_cast<uint4*>(&As[buf ^ 1][tid * LDS_STRIDE]);
            dA[0] = ra0; dA[1] = ra1; dA[2] = ra2; dA[3] = ra3;
            uint4* dB = reinterpret_cast<uint4*>(&Bs[buf ^ 1][tid * LDS_STRIDE]);
            dB[0] = rb0; dB[1] = rb1; dB[2] = rb2; dB[3] = rb3;
            __syncthreads();
            buf ^= 1;
        }
    }

    // ---- epilogue ----
    #pragma unroll
    for (int i = 0; i < 4; i++) {
        #pragma unroll
        for (int j = 0; j < 4; j++) {
            int gmBase = m0 + wm * 64 + i * 16 + (g << 3);
            int gn     = n0 + wn * 64 + j * 16 + mloc;
            float bval = bias[gn];
            #pragma unroll
            for (int r = 0; r < 8; r++) {
                int   gm  = gmBase + r;            // m = r + 8g
                float val = acc[i][j][r] + bval;
                if (mode == 0) {
                    outF[(size_t)gm * N + gn] = val;
                } else {
                    int b  = gm >> 11;             // /T
                    int t  = gm & (Tv - 1);
                    int hh = gn >> 6;              // /HD
                    int d  = gn & (HDv - 1);
                    if (mode == 1) {
                        val *= 0.125f;             // 1/sqrt(HD)
                        outH[(((size_t)(b * NHv + hh) * Tv) + t) * HDv + d] = (half_t)val;
                    } else if (mode == 2) {
                        outH[(((size_t)(b * NHv + hh) * Tv) + t) * HDv + d] = (half_t)val;
                    } else {
                        outH[(((size_t)(b * NHv + hh) * HDv) + d) * Tv + t] = (half_t)val;
                    }
                }
            }
        }
    }
}

// ---------------------------------------------------------------------------
// Causal flash attention. One wave per 32-row q tile (two 16-row m-tiles
// sharing K/V fragments), k-blocks of 32 cols.
// Q: [B,NH,T,HD] f16 (pre-scaled), K: [B,NH,T,HD] f16, Vt: [B,NH,HD,T] f16.
// Y: [B,T,C] f16.
// ---------------------------------------------------------------------------
#define PS_STRIDE 34   // halves per row of P scratch (32 + 2 pad)

__launch_bounds__(128, 1)
__global__ void attn_kernel(const half_t* __restrict__ Q,
                            const half_t* __restrict__ Km,
                            const half_t* __restrict__ Vt,
                            half_t* __restrict__ Y)
{
    __shared__ unsigned short Ps[4][32 * PS_STRIDE];

    const int tid  = threadIdx.x;
    const int wave = tid >> 5;
    const int lane = tid & 31;
    const int mloc = lane & 15;
    const int g    = lane >> 4;
    const int bh   = blockIdx.y;                     // b*NH + h
    const int qt   = (blockIdx.x * 4 + wave) * 32;   // q tile start row (32 rows)

    const half_t* qp = Q  + (size_t)bh * Tv * HDv;
    const half_t* kp = Km + (size_t)bh * Tv * HDv;
    const half_t* vp = Vt + (size_t)bh * HDv * Tv;
    unsigned short* myPs = &Ps[wave][0];

    // Q A-fragments: qf[u][h] = rows qt+16u.., d-chunk 32h..
    AFrag qf[2][2];
    #pragma unroll
    for (int u = 0; u < 2; u++) {
        const unsigned int* qrow =
            reinterpret_cast<const unsigned int*>(qp + (size_t)(qt + u * 16 + mloc) * HDv);
        #pragma unroll
        for (int hc = 0; hc < 2; hc++) {
            #pragma unroll
            for (int v = 0; v < 8; v++) {
                int kk = hc * 32 + (g << 4) + (v << 1);
                qf[u][hc].u[v] = qrow[kk >> 1];
            }
        }
    }

    v8f o[2][4] = {};
    float runMax[2][8], Lp[2][8];
    #pragma unroll
    for (int u = 0; u < 2; u++)
        #pragma unroll
        for (int r = 0; r < 8; r++) { runMax[u][r] = -1e30f; Lp[u][r] = 0.0f; }

    for (int j0 = 0; j0 <= qt; j0 += 32) {
        const bool diag = (j0 + 32 > qt);

        // ---- S = Q * K^T: K fragments shared by both m-tiles ----
        v8f s[2][2];
        #pragma unroll
        for (int c = 0; c < 2; c++) {
            AFrag kf0, kf1;
            const unsigned int* krow = reinterpret_cast<const unsigned int*>(
                kp + (size_t)(j0 + c * 16 + mloc) * HDv);
            #pragma unroll
            for (int v = 0; v < 8; v++) {
                kf0.u[v] = krow[(g << 3) + v];        // d = 16g + 2v (+1)
                kf1.u[v] = krow[16 + (g << 3) + v];   // d = 32 + 16g + 2v (+1)
            }
            #pragma unroll
            for (int u = 0; u < 2; u++) {
                v8f sc = {};
                sc = __builtin_amdgcn_wmma_f32_16x16x32_f16(false, qf[u][0].h, false, kf0.h,
                                                            (short)0, sc, false, false);
                sc = __builtin_amdgcn_wmma_f32_16x16x32_f16(false, qf[u][1].h, false, kf1.h,
                                                            (short)0, sc, false, false);
                s[u][c] = sc;
            }
        }

        if (diag) {
            #pragma unroll
            for (int u = 0; u < 2; u++)
                #pragma unroll
                for (int c = 0; c < 2; c++)
                    #pragma unroll
                    for (int r = 0; r < 8; r++) {
                        int row = qt + u * 16 + r + (g << 3);
                        int col = j0 + c * 16 + mloc;
                        if (col > row) s[u][c][r] = -1e30f;
                    }
        }

        // ---- online softmax per m-tile ----
        #pragma unroll
        for (int u = 0; u < 2; u++) {
            float alpha[8];
            #pragma unroll
            for (int r = 0; r < 8; r++) {
                float m = fmaxf(s[u][0][r], s[u][1][r]);
                m = fmaxf(m, __shfl_xor(m, 1, 32));
                m = fmaxf(m, __shfl_xor(m, 2, 32));
                m = fmaxf(m, __shfl_xor(m, 4, 32));
                m = fmaxf(m, __shfl_xor(m, 8, 32));   // reduce within 16-lane group
                float nm = fmaxf(runMax[u][r], m);
                alpha[r] = __expf(runMax[u][r] - nm);
                runMax[u][r] = nm;
            }
            #pragma unroll
            for (int r = 0; r < 8; r++) {
                float p0 = __expf(s[u][0][r] - runMax[u][r]);
                float p1 = __expf(s[u][1][r] - runMax[u][r]);
                Lp[u][r] = Lp[u][r] * alpha[r] + p0 + p1;
                int row = u * 16 + r + (g << 3);
                myPs[row * PS_STRIDE + mloc]      = __builtin_bit_cast(unsigned short, (half_t)p0);
                myPs[row * PS_STRIDE + 16 + mloc] = __builtin_bit_cast(unsigned short, (half_t)p1);
                #pragma unroll
                for (int t = 0; t < 4; t++) o[u][t][r] *= alpha[r];
            }
        }

        // ---- P C-frag -> A-frag via per-wave LDS scratch ----
        AFrag pf[2];
        #pragma unroll
        for (int u = 0; u < 2; u++)
            #pragma unroll
            for (int i = 0; i < 16; i++) {
                int v = i >> 1, pbit = i & 1;
                int kk = (g << 4) + (v << 1) + pbit;
                pf[u].s[i] = myPs[(u * 16 + mloc) * PS_STRIDE + kk];
            }

        // ---- O += P * V: V fragments shared by both m-tiles ----
        #pragma unroll
        for (int t = 0; t < 4; t++) {
            AFrag vf;
            const unsigned int* vrow = reinterpret_cast<const unsigned int*>(
                vp + (size_t)(t * 16 + mloc) * Tv + j0);
            #pragma unroll
            for (int v = 0; v < 8; v++) vf.u[v] = vrow[(g << 3) + v];
            #pragma unroll
            for (int u = 0; u < 2; u++)
                o[u][t] = __builtin_amdgcn_wmma_f32_16x16x32_f16(false, pf[u].h, false, vf.h,
                                                                 (short)0, o[u][t], false, false);
        }
    }

    // ---- normalize and write y = [B,T,C] f16 ----
    const int hh = bh & (NHv - 1);
    const int bb = bh >> 4;
    #pragma unroll
    for (int u = 0; u < 2; u++) {
        #pragma unroll
        for (int r = 0; r < 8; r++) {
            float l = Lp[u][r];
            l += __shfl_xor(l, 1, 32);
            l += __shfl_xor(l, 2, 32);
            l += __shfl_xor(l, 4, 32);
            l += __shfl_xor(l, 8, 32);
            float inv = 1.0f / l;
            int row = qt + u * 16 + r + (g << 3);
            #pragma unroll
            for (int t = 0; t < 4; t++) {
                int col = hh * HDv + t * 16 + mloc;
                Y[((size_t)bb * Tv + row) * Cv + col] = (half_t)(o[u][t][r] * inv);
            }
        }
    }
}

// ---------------------------------------------------------------------------
extern "C" void kernel_launch(void* const* d_in, const int* in_sizes, int n_in,
                              void* d_out, int out_size, void* d_ws, size_t ws_size,
                              hipStream_t stream)
{
    (void)in_sizes; (void)n_in; (void)out_size; (void)ws_size;

    const float* x  = (const float*)d_in[0];
    const float* Wq = (const float*)d_in[1];
    const float* bq = (const float*)d_in[2];
    const float* Wk = (const float*)d_in[3];
    const float* bk = (const float*)d_in[4];
    const float* Wv = (const float*)d_in[5];
    const float* bv = (const float*)d_in[6];
    const float* Wp = (const float*)d_in[7];
    const float* bp = (const float*)d_in[8];
    float* out = (float*)d_out;

    const size_t NX = (size_t)Mtot * Cv;   // 8,388,608
    const size_t NW = (size_t)Cv * Cv;     // 1,048,576

    half_t* ws  = (half_t*)d_ws;
    half_t* xh  = ws;                size_t off = NX;
    half_t* Wqh = ws + off;          off += NW;
    half_t* Wkh = ws + off;          off += NW;
    half_t* Wvh = ws + off;          off += NW;
    half_t* Wph = ws + off;          off += NW;
    half_t* qh  = ws + off;          off += NX;
    half_t* kh  = ws + off;          off += NX;
    half_t* vTh = ws + off;          off += NX;
    half_t* yh  = ws + off;          off += NX;

    // 1) f32 -> f16 conversions
    cvt_f32_f16<<<(int)(NX / (4 * 256)), 256, 0, stream>>>(x,  xh,  (int)NX);
    cvt_f32_f16<<<(int)(NW / (4 * 256)), 256, 0, stream>>>(Wq, Wqh, (int)NW);
    cvt_f32_f16<<<(int)(NW / (4 * 256)), 256, 0, stream>>>(Wk, Wkh, (int)NW);
    cvt_f32_f16<<<(int)(NW / (4 * 256)), 256, 0, stream>>>(Wv, Wvh, (int)NW);
    cvt_f32_f16<<<(int)(NW / (4 * 256)), 256, 0, stream>>>(Wp, Wph, (int)NW);

    // 2) QKV projections (x @ W^T + b) with layout-scattering epilogues
    dim3 ggrid(Cv / 128, Mtot / 128);   // (8, 64)
    gemm_wmma<<<ggrid, 128, 0, stream>>>(xh, Wqh, bq, qh,  nullptr, Mtot, Cv, Cv, 1);
    gemm_wmma<<<ggrid, 128, 0, stream>>>(xh, Wkh, bk, kh,  nullptr, Mtot, Cv, Cv, 2);
    gemm_wmma<<<ggrid, 128, 0, stream>>>(xh, Wvh, bv, vTh, nullptr, Mtot, Cv, Cv, 3);

    // 3) causal flash attention (32 q rows per wave)
    dim3 agrid(Tv / 128, Bv * NHv);     // (16, 64)
    attn_kernel<<<agrid, 128, 0, stream>>>(qh, kh, vTh, yh);

    // 4) output projection (f32 out + bias)
    gemm_wmma<<<ggrid, 128, 0, stream>>>(yh, Wph, bp, nullptr, out, Mtot, Cv, Cv, 0);
}
